// MultiHeadAttention_77893526880319
// MI455X (gfx1250) — matrix operand order repaired and checked by reading, not compiled, not run on hardware
//
#include <hip/hip_runtime.h>

#define N_HEAD 16
#define D_MODEL 1024
#define BATCH 2
#define SEQ 2048
#define DH 64
#define M_ROWS (BATCH * SEQ)   // 4096

typedef __bf16 bf16;
typedef __attribute__((ext_vector_type(16))) __bf16 v16bf;
typedef __attribute__((ext_vector_type(8)))  __bf16 v8bf;
typedef __attribute__((ext_vector_type(8)))  float  v8f;

union FragA { v16bf v; v8bf h[2]; };

// ---------------------------------------------------------------------------
// Async global->LDS copy (ASYNCcnt path) with graceful fallback.
// Builtin signature (per hipcc diagnostic): param0 is a pointer to a 16-byte
// int vector in the global address space; param1 the LDS-side equivalent.
// ---------------------------------------------------------------------------
#if __has_builtin(__builtin_amdgcn_global_load_async_to_lds_b128)
#define ASYNC_LDS 1
#else
#define ASYNC_LDS 0
#endif

typedef int v4i32 __attribute__((vector_size(16)));
typedef __attribute__((address_space(1))) v4i32* as1_v4i32_p;
typedef __attribute__((address_space(3))) v4i32* as3_v4i32_p;

__device__ __forceinline__ void async_copy_b128(const bf16* g, bf16* l) {
#if ASYNC_LDS
  __builtin_amdgcn_global_load_async_to_lds_b128((as1_v4i32_p)g,
                                                 (as3_v4i32_p)l, 0, 0);
#else
  *(v8bf*)l = *(const v8bf*)g;
#endif
}

__device__ __forceinline__ void wait_async() {
#if ASYNC_LDS
#if __has_builtin(__builtin_amdgcn_s_wait_asynccnt)
  __builtin_amdgcn_s_wait_asynccnt(0);
#else
  asm volatile("s_wait_asynccnt 0" ::: "memory");
#endif
#endif
}

// ---------------------------------------------------------------------------
// Pure-VALU reductions across each 16-lane half (DPP, no LDS / bpermute).
// Steps: xor1 (quad_perm [1,0,3,2]), xor2 (quad_perm [2,3,0,1]); after those
// values are uniform per 4-group so row_half_mirror acts as xor4 and
// row_mirror acts as xor8.
// ---------------------------------------------------------------------------
template <int CTRL>
__device__ __forceinline__ float dpp_f(float x) {
  return __builtin_bit_cast(
      float, __builtin_amdgcn_update_dpp(0, __builtin_bit_cast(int, x), CTRL,
                                         0xf, 0xf, true));
}
__device__ __forceinline__ float red16_max(float v) {
  v = fmaxf(v, dpp_f<0xB1>(v));    // quad_perm [1,0,3,2]
  v = fmaxf(v, dpp_f<0x4E>(v));    // quad_perm [2,3,0,1]
  v = fmaxf(v, dpp_f<0x141>(v));   // row_half_mirror
  v = fmaxf(v, dpp_f<0x140>(v));   // row_mirror
  return v;
}
__device__ __forceinline__ float red16_sum(float v) {
  v += dpp_f<0xB1>(v);
  v += dpp_f<0x4E>(v);
  v += dpp_f<0x141>(v);
  v += dpp_f<0x140>(v);
  return v;
}

// ---------------------------------------------------------------------------
// fp32 -> bf16 cast (grid-stride)
// ---------------------------------------------------------------------------
__global__ void cast_bf16_kernel(const float* __restrict__ src,
                                 bf16* __restrict__ dst, int n) {
  int i = blockIdx.x * blockDim.x + threadIdx.x;
  int stride = gridDim.x * blockDim.x;
  for (; i < n; i += stride) dst[i] = (bf16)src[i];
}

// ---------------------------------------------------------------------------
// C[M,N] = A[M,K] @ W[N,K]^T + bias.  One 16x64 output strip per wave:
// A fragment loaded once per k-step, reused by 4 B fragments / 4 WMMAs.
// ---------------------------------------------------------------------------
template <bool OUT_F32>
__global__ __launch_bounds__(256)
void gemm_nt_wmma(const bf16* __restrict__ A, const bf16* __restrict__ W,
                  const float* __restrict__ bias, void* __restrict__ Cout,
                  int M, int N, int K) {
  const int lane   = threadIdx.x & 31;
  const int wave   = threadIdx.x >> 5;
  const int waveId = blockIdx.x * 8 + wave;
  const int nblk   = N >> 6;            // 64-wide N strips
  const int mtile  = waveId / nblk;
  const int nstrip = waveId % nblk;
  if (mtile >= (M >> 4)) return;

  const int row = lane & 15;
  const int hi  = lane >> 4;
  const int m0  = mtile * 16;
  const int n0  = nstrip * 64;

  v8f acc[4] = {{}, {}, {}, {}};
  for (int k0 = 0; k0 < K; k0 += 32) {
    FragA a;
    const bf16* ap = A + (size_t)(m0 + row) * K + k0 + hi * 8;
    a.h[0] = *(const v8bf*)(ap);
    a.h[1] = *(const v8bf*)(ap + 16);
#pragma unroll
    for (int nt = 0; nt < 4; ++nt) {
      FragA b;
      const bf16* bp = W + (size_t)(n0 + nt * 16 + row) * K + k0 + hi * 8;
      b.h[0] = *(const v8bf*)(bp);
      b.h[1] = *(const v8bf*)(bp + 16);
      acc[nt] = __builtin_amdgcn_wmma_f32_16x16x32_bf16(
          false, a.v, false, b.v, (short)0, acc[nt], false, false);
    }
  }

#pragma unroll
  for (int nt = 0; nt < 4; ++nt) {
    const int n = n0 + nt * 16 + row;
    const float bn = bias[n];
#pragma unroll
    for (int r = 0; r < 8; ++r) {
      float v = acc[nt][r] + bn;
      int m = m0 + r + hi * 8;
      if (OUT_F32)
        ((float*)Cout)[(size_t)m * N + n] = v;
      else
        ((bf16*)Cout)[(size_t)m * N + n] = (bf16)v;
    }
  }
}

// ---------------------------------------------------------------------------
// Vt[b,h,d,t] = Vw[b,t,h*DH+d]
// ---------------------------------------------------------------------------
__global__ void transpose_v_kernel(const bf16* __restrict__ Vw,
                                   bf16* __restrict__ Vt) {
  size_t i = (size_t)blockIdx.x * blockDim.x + threadIdx.x;
  size_t total = (size_t)BATCH * SEQ * D_MODEL;
  if (i >= total) return;
  int d = (int)(i % DH);
  int h = (int)((i / DH) % N_HEAD);
  int t = (int)((i / D_MODEL) % SEQ);
  int b = (int)(i / ((size_t)SEQ * D_MODEL));
  Vt[(((size_t)b * N_HEAD + h) * DH + d) * SEQ + t] = Vw[i];
}

// ---------------------------------------------------------------------------
// Causal flash attention.  One workgroup (8 waves) owns (b, h) and 128 query
// rows; per 32-key chunk the whole WG async-stages the K tile (32x64) and the
// Vt tile (64x32) into LDS, then each wave does: 2x2 chained score WMMAs,
// online softmax (DPP reductions), P relayout via wave-private LDS, 4 PV
// WMMAs.  The t-loop bound is block-uniform so barriers stay convergent;
// waves whose chunk is fully masked skip compute.
// ---------------------------------------------------------------------------
__global__ __launch_bounds__(256)
void flash_attn_kernel(const bf16* __restrict__ Qw, const bf16* __restrict__ Kw,
                       const bf16* __restrict__ Vt, bf16* __restrict__ Out) {
  __shared__ __align__(16) bf16 kTile[32 * 64];      // [t_local][dh]
  __shared__ __align__(16) bf16 vTile[64 * 32];      // [d][t_local]
  __shared__ __align__(16) bf16 pAll[8][16 * 32];    // per-wave P tile

  const int tid  = threadIdx.x;
  const int lane = tid & 31;
  const int wave = tid >> 5;
  bf16* plds = pAll[wave];

  const int mblocks = SEQ / 128;                 // 16
  const int mblk = blockIdx.x % mblocks;
  const int bh   = blockIdx.x / mblocks;
  const int h    = bh % N_HEAD;
  const int b    = bh / N_HEAD;
  const int M0   = mblk * 128;
  const int m0   = M0 + wave * 16;               // this wave's query tile
  const int row  = lane & 15;
  const int hi   = lane >> 4;

  // Q fragments (loop invariant): two 32-wide K-chunks over Dh=64
  FragA q[2];
  const bf16* qbase = Qw + ((size_t)b * SEQ + m0 + row) * D_MODEL + h * DH;
#pragma unroll
  for (int c = 0; c < 2; ++c) {
    q[c].h[0] = *(const v8bf*)(qbase + c * 32 + hi * 8);
    q[c].h[1] = *(const v8bf*)(qbase + c * 32 + hi * 8 + 16);
  }

  const bf16* kbase = Kw + (size_t)b * SEQ * D_MODEL + h * DH;
  const bf16* vbase = Vt + ((size_t)b * N_HEAD + h) * DH * SEQ;

  // cooperative staging map: 256 threads x 16B
  const int krow = tid >> 3, kseg = tid & 7;     // K: 32 rows x 8 segs
  const int vrow = tid >> 2, vseg = tid & 3;     // V: 64 rows x 4 segs

  const float scale = 1.0f / 32.0f;              // 1/sqrt(d_model)
  float mrow[8], lrow[8];
#pragma unroll
  for (int r = 0; r < 8; ++r) { mrow[r] = -3.0e38f; lrow[r] = 0.0f; }
  v8f o[4] = {{}, {}, {}, {}};

  const int tEnd = M0 + 128;                     // block-uniform
  for (int t0 = 0; t0 < tEnd; t0 += 32) {
    __syncthreads();   // WAR: previous chunk's LDS reads complete
    async_copy_b128(kbase + (size_t)(t0 + krow) * D_MODEL + kseg * 8,
                    kTile + krow * 64 + kseg * 8);
    async_copy_b128(vbase + (size_t)vrow * SEQ + t0 + vseg * 8,
                    vTile + vrow * 32 + vseg * 8);
    wait_async();
    __syncthreads();   // tiles visible to all waves

    if (t0 <= m0 + 15) {   // chunk has at least one unmasked column
      // ---- scores: two 16-column subtiles from LDS K tile
      v8f s[2];
#pragma unroll
      for (int ns = 0; ns < 2; ++ns) {
        const bf16* kp = kTile + (ns * 16 + row) * 64;
        FragA kb0, kb1;
        kb0.h[0] = *(const v8bf*)(kp + hi * 8);
        kb0.h[1] = *(const v8bf*)(kp + hi * 8 + 16);
        kb1.h[0] = *(const v8bf*)(kp + 32 + hi * 8);
        kb1.h[1] = *(const v8bf*)(kp + 32 + hi * 8 + 16);
        v8f c = {};
        c = __builtin_amdgcn_wmma_f32_16x16x32_bf16(false, q[0].v, false, kb0.v,
                                                    (short)0, c, false, false);
        c = __builtin_amdgcn_wmma_f32_16x16x32_bf16(false, q[1].v, false, kb1.v,
                                                    (short)0, c, false, false);
        s[ns] = c;
      }

      // ---- scale + causal mask
#pragma unroll
      for (int ns = 0; ns < 2; ++ns) {
        const int tcol = t0 + ns * 16 + row;
#pragma unroll
        for (int r = 0; r < 8; ++r) {
          int m = m0 + r + hi * 8;
          float v = s[ns][r] * scale;
          s[ns][r] = (tcol > m) ? -3.0e38f : v;
        }
      }

      // ---- online softmax (DPP half-wave reductions)
      float alpha[8];
#pragma unroll
      for (int r = 0; r < 8; ++r) {
        float v = red16_max(fmaxf(s[0][r], s[1][r]));
        float mn = fmaxf(mrow[r], v);
        alpha[r] = __expf(mrow[r] - mn);
        mrow[r] = mn;
      }
#pragma unroll
      for (int ns = 0; ns < 2; ++ns)
#pragma unroll
        for (int r = 0; r < 8; ++r) s[ns][r] = __expf(s[ns][r] - mrow[r]);
#pragma unroll
      for (int r = 0; r < 8; ++r) {
        float ps = red16_sum(s[0][r] + s[1][r]);
        lrow[r] = lrow[r] * alpha[r] + ps;
      }
#pragma unroll
      for (int ds = 0; ds < 4; ++ds)
#pragma unroll
        for (int r = 0; r < 8; ++r) o[ds][r] *= alpha[r];

      // ---- P: C/D layout -> A-operand layout via wave-private LDS tile
#pragma unroll
      for (int ns = 0; ns < 2; ++ns)
#pragma unroll
        for (int r = 0; r < 8; ++r)
          plds[(r + hi * 8) * 32 + ns * 16 + row] = (bf16)s[ns][r];

      FragA p;
      {
        const bf16* pp = plds + row * 32 + hi * 8;
        p.h[0] = *(const v8bf*)(pp);
        p.h[1] = *(const v8bf*)(pp + 16);
      }

      // ---- P @ V over this chunk, 4 d-subtiles
#pragma unroll
      for (int ds = 0; ds < 4; ++ds) {
        FragA vb;
        const bf16* vp = vTile + (ds * 16 + row) * 32 + hi * 8;
        vb.h[0] = *(const v8bf*)(vp);
        vb.h[1] = *(const v8bf*)(vp + 16);
        o[ds] = __builtin_amdgcn_wmma_f32_16x16x32_bf16(
            false, p.v, false, vb.v, (short)0, o[ds], false, false);
      }
    }
  }

  // ---- epilogue: normalize, write bf16 [B,S,D]
  bf16* obase = Out + (size_t)b * SEQ * D_MODEL + h * DH;
#pragma unroll
  for (int ds = 0; ds < 4; ++ds) {
    int d = ds * 16 + row;
#pragma unroll
    for (int r = 0; r < 8; ++r) {
      int m = m0 + r + hi * 8;
      obase[(size_t)m * D_MODEL + d] = (bf16)(o[ds][r] / lrow[r]);
    }
  }
}

// ---------------------------------------------------------------------------
extern "C" void kernel_launch(void* const* d_in, const int* in_sizes, int n_in,
                              void* d_out, int out_size, void* d_ws, size_t ws_size,
                              hipStream_t stream) {
  const float* q  = (const float*)d_in[0];
  const float* k  = (const float*)d_in[1];
  const float* v  = (const float*)d_in[2];
  const float* Wq = (const float*)d_in[3];
  const float* bq = (const float*)d_in[4];
  const float* Wk = (const float*)d_in[5];
  const float* bk = (const float*)d_in[6];
  const float* Wv = (const float*)d_in[7];
  const float* bv = (const float*)d_in[8];
  const float* Wo = (const float*)d_in[9];
  const float* bo = (const float*)d_in[10];

  const size_t X   = (size_t)M_ROWS * D_MODEL;   // 4 Mi elements
  const size_t WSZ = (size_t)D_MODEL * D_MODEL;  // 1 Mi elements

  bf16* ws  = (bf16*)d_ws;
  bf16* Xq  = ws;
  bf16* Xk  = Xq + X;
  bf16* Xv  = Xk + X;
  bf16* Bwq = Xv + X;
  bf16* Bwk = Bwq + WSZ;
  bf16* Bwv = Bwk + WSZ;
  bf16* Bwo = Bwv + WSZ;
  bf16* Qw  = Bwo + WSZ;
  bf16* Kw  = Qw + X;
  bf16* Vw  = Kw + X;
  bf16* Vtp = Vw + X;
  bf16* At  = Vtp + X;

  // casts to bf16
  cast_bf16_kernel<<<2048, 256, 0, stream>>>(q,  Xq,  (int)X);
  cast_bf16_kernel<<<2048, 256, 0, stream>>>(k,  Xk,  (int)X);
  cast_bf16_kernel<<<2048, 256, 0, stream>>>(v,  Xv,  (int)X);
  cast_bf16_kernel<<<1024, 256, 0, stream>>>(Wq, Bwq, (int)WSZ);
  cast_bf16_kernel<<<1024, 256, 0, stream>>>(Wk, Bwk, (int)WSZ);
  cast_bf16_kernel<<<1024, 256, 0, stream>>>(Wv, Bwv, (int)WSZ);
  cast_bf16_kernel<<<1024, 256, 0, stream>>>(Wo, Bwo, (int)WSZ);

  // projections: (M/16)*(N/64) waves / 8 per block = 512 blocks
  const int gemmBlocks = (M_ROWS / 16) * (D_MODEL / 64) / 8;
  gemm_nt_wmma<false><<<gemmBlocks, 256, 0, stream>>>(Xq, Bwq, bq, Qw,
                                                      M_ROWS, D_MODEL, D_MODEL);
  gemm_nt_wmma<false><<<gemmBlocks, 256, 0, stream>>>(Xk, Bwk, bk, Kw,
                                                      M_ROWS, D_MODEL, D_MODEL);
  gemm_nt_wmma<false><<<gemmBlocks, 256, 0, stream>>>(Xv, Bwv, bv, Vw,
                                                      M_ROWS, D_MODEL, D_MODEL);

  transpose_v_kernel<<<(int)((X + 255) / 256), 256, 0, stream>>>(Vw, Vtp);

  // flash attention: B*H*(S/128) = 512 workgroups of 8 waves
  flash_attn_kernel<<<512, 256, 0, stream>>>(Qw, Kw, Vtp, At);

  // output projection (fp32 + bias) straight to d_out
  gemm_nt_wmma<true><<<gemmBlocks, 256, 0, stream>>>(At, Bwo, bo, (float*)d_out,
                                                     M_ROWS, D_MODEL, D_MODEL);
}
